// MSSFM_13529146982479
// MI455X (gfx1250) — matrix operand order repaired and checked by reading, not compiled, not run on hardware
//
#include <hip/hip_runtime.h>
#include <hip/hip_bf16.h>

// ---------------------------------------------------------------------------
// MSSFM block for MI455X (gfx1250, wave32, WMMA, TDM).
// All intermediates stored NHWC ([B][HW][C], channels contiguous) so that
// GEMM tiles are contiguous 2-D blocks -> loaded by the Tensor Data Mover
// with hardware row padding (conflict-free LDS), and epilogue stores are
// 16-byte contiguous per lane. Final output converted back to NCHW.
//
// Pipeline:
//   0) cvt kernels     : 1x1-conv weights f32 -> bf16 (once, ~0.3 MB)
//   1) conv_dpb_kernel : dual dilated 3x3 conv via im2col WMMA (K=96)
//                        + PReLU + sum -> y1 (f32 NHWC), y1b (bf16 NHWC)
//   2-5) gemm_bias_act : 1x1 GEMMs (bf16 WMMA, f32 acc, TDM-staged tiles)
//                        fused tanh/sigmoid -> t, Z, Fg
//   6) scan_gate_kernel: SRU scan over H (fwd/rev split wave-uniform),
//                        fused gate ygate = y1 * sigmoid(y2)
//   7-9) gemm_bias_act : PSA (v*y1+y1), GELU(pw1), final pw2 GEMM with
//                        epilogue out = v + ygate -> d_out (NCHW)
// Workspace: ~161 MB.
// ---------------------------------------------------------------------------

typedef __attribute__((ext_vector_type(16))) __bf16       v16bf;
typedef __attribute__((ext_vector_type(8)))  float        v8f;
typedef __attribute__((ext_vector_type(4)))  unsigned int u32x4;

union Frag {
    u32x4 u[2];   // 32 bytes = 16 bf16
    v16bf v;
};

#define HWC   16384   // H*W
#define IMG_W 128
#define IMG_H 128
#define NB    4

__device__ __forceinline__ unsigned short f2bf_bits(float f) {
    __hip_bfloat16 h = __float2bfloat16(f);
    unsigned short s;
    __builtin_memcpy(&s, &h, 2);
    return s;
}

// ---------------------------------------------------------------------------
// Tensor Data Mover: 2-D tile load, bf16 elements, 16-byte pad per row so the
// LDS image has row stride K+8 bf16 (bank-conflict-free fragment reads).
// D# layout per CDNA5 ISA 8.3/8.4. Issued wave-uniform by wave 0 only.
// This toolchain exposes the 6-arg builtin:
//   (u32x4 g0, i32x8 g1, i32x4 g2, i32x4 g3, i32x8 g4, i32 cpol)
// ---------------------------------------------------------------------------
#if __has_builtin(__builtin_amdgcn_tensor_load_to_lds)
#define HAVE_TDM 1
typedef unsigned int tdm_u32x4 __attribute__((ext_vector_type(4)));
typedef int          tdm_i32x8 __attribute__((ext_vector_type(8)));
typedef int          tdm_i32x4 __attribute__((ext_vector_type(4)));

template<int K, int ROWS>
__device__ __forceinline__ void tdm_load_tile_pad16(const void* gptr,
                                                    unsigned lds_off) {
    // pad_interval code: 2^(code+1) DWORDs before pad; row = K*2B = K/2 DWORDs
    constexpr int PIC = (K == 128) ? 5 : 6;       // 64 dw / 128 dw rows
    static_assert(K == 128 || K == 256, "row must be 2^n dwords for padding");
    unsigned long long ga = (unsigned long long)gptr;
    tdm_u32x4 g0;
    g0[0] = 1u;                                   // count=1, user descriptor
    g0[1] = lds_off;                              // LDS byte address
    g0[2] = (unsigned)(ga & 0xFFFFFFFFu);         // global_addr[31:0]
    g0[3] = (unsigned)((ga >> 32) & 0x1FFFFFFu)   // global_addr[56:32]
          | (2u << 30);                           // type = 2 ("image")
    tdm_i32x8 g1;
    g1[0] = (1 << 16)                             // data_size = 2 bytes
          | (1 << 20)                             // pad_enable
          | (PIC << 22)                           // pad_interval
          | (3 << 25);                            // pad_amount = 4 dw = 16 B
    g1[1] = (K & 0xFFFF) << 16;                   // tensor_dim0[15:0]
    g1[2] = ((unsigned)K >> 16)                   // tensor_dim0[31:16]
          | ((ROWS & 0xFFFF) << 16);              // tensor_dim1[15:0]
    g1[3] = ((unsigned)ROWS >> 16)                // tensor_dim1[31:16]
          | ((K & 0xFFFF) << 16);                 // tile_dim0 = K
    g1[4] = (ROWS & 0xFFFF);                      // tile_dim1; tile_dim2 = 0
    g1[5] = K;                                    // tensor_dim0_stride[31:0]
    g1[6] = 0;
    g1[7] = 0;
    tdm_i32x4 z4 = {0, 0, 0, 0};
    tdm_i32x8 z8 = {0, 0, 0, 0, 0, 0, 0, 0};
    __builtin_amdgcn_tensor_load_to_lds(g0, g1, z4, z4, z8, 0);
}
#endif

// ---------------------------------------------------------------------------
// 0) weight conversion f32 -> bf16
// ---------------------------------------------------------------------------
__global__ __launch_bounds__(256)
void cvt_f32_bf16_kernel(const float* __restrict__ src,
                         __hip_bfloat16* __restrict__ dst, int n) {
    int i = blockIdx.x * 256 + threadIdx.x;
    if (i < n) dst[i] = __float2bfloat16(src[i]);
}

// ---------------------------------------------------------------------------
// 1) Dual-dilation 3x3 conv (S=8 -> C=128), PReLU both branches, sum.
//    im2col GEMM: M=128, K=72 pad 96, N=64 pixels/block (half an image row).
//    Outputs NHWC: y1 f32, y1b bf16.
// ---------------------------------------------------------------------------
__global__ __launch_bounds__(256)
void conv_dpb_kernel(const float* __restrict__ x,     // [B][8][128][128] NCHW
                     const float* __restrict__ w1,    // [128][8][3][3]
                     const float* __restrict__ b1,
                     const float* __restrict__ w4,
                     const float* __restrict__ b4,
                     const float* __restrict__ pa,
                     float* __restrict__ y1,          // [B][HW][128] f32
                     __hip_bfloat16* __restrict__ y1b)// [B][HW][128] bf16
{
    constexpr int KP = 104;                           // 96 + 8 pad (bf16)
    extern __shared__ char smem[];
    __hip_bfloat16* wl = (__hip_bfloat16*)smem;       // [2][128][KP]
    __hip_bfloat16* im = wl + 2 * 128 * KP;           // [2][64][KP]

    const int tid = threadIdx.x;
    const int b   = blockIdx.y;
    const int n0  = blockIdx.x * 64;
    const int h   = n0 >> 7;
    const int w0c = n0 & 127;

    for (int idx = tid; idx < 2 * 128 * 96; idx += 256) {
        int br = idx / (128 * 96);
        int r  = idx % (128 * 96);
        int co = r / 96;
        int k  = r % 96;
        float v = 0.f;
        if (k < 72) v = (br ? w4 : w1)[co * 72 + k];
        wl[(br * 128 + co) * KP + k] = __float2bfloat16(v);
    }
    for (int idx = tid; idx < 2 * 64 * 96; idx += 256) {
        int br = idx / (64 * 96);
        int r  = idx % (64 * 96);
        int p  = r / 96;
        int k  = r % 96;
        float v = 0.f;
        if (k < 72) {
            int s  = k / 9;
            int t  = k % 9;
            int d  = br ? 7 : 1;
            int hh = h + d * ((t / 3) - 1);
            int ww = w0c + p + d * ((t % 3) - 1);
            if (hh >= 0 && hh < IMG_H && ww >= 0 && ww < IMG_W)
                v = x[(((size_t)b * 8 + s) * IMG_H + hh) * IMG_W + ww];
        }
        im[(br * 64 + p) * KP + k] = __float2bfloat16(v);
    }
    __syncthreads();

    const int lane   = tid & 31;
    const int waveId = tid >> 5;
    const int wm     = (waveId & 3) * 32;
    const int wn     = (waveId >> 2) * 32;
    const int l16    = lane & 15;
    const int lhi    = lane >> 4;

    v8f acc[2][2][2] = {};
    #pragma unroll
    for (int k0 = 0; k0 < 96; k0 += 32) {
        Frag a[2][2], bf[2][2];
        #pragma unroll
        for (int br = 0; br < 2; ++br) {
            #pragma unroll
            for (int i = 0; i < 2; ++i) {
                const __hip_bfloat16* p =
                    &wl[(br * 128 + wm + i * 16 + l16) * KP + k0 + lhi * 8];
                a[br][i].u[0] = *(const u32x4*)(p);
                a[br][i].u[1] = *(const u32x4*)(p + 16);
            }
            #pragma unroll
            for (int j = 0; j < 2; ++j) {
                const __hip_bfloat16* p =
                    &im[(br * 64 + wn + j * 16 + l16) * KP + k0 + lhi * 8];
                bf[br][j].u[0] = *(const u32x4*)(p);
                bf[br][j].u[1] = *(const u32x4*)(p + 16);
            }
        }
        #pragma unroll
        for (int br = 0; br < 2; ++br)
            #pragma unroll
            for (int i = 0; i < 2; ++i)
                #pragma unroll
                for (int j = 0; j < 2; ++j)
                    acc[br][i][j] = __builtin_amdgcn_wmma_f32_16x16x32_bf16(
                        false, a[br][i].v, false, bf[br][j].v,
                        (short)0, acc[br][i][j], false, false);
    }

    const float alpha = pa[0];
    #pragma unroll
    for (int i = 0; i < 2; ++i) {
        #pragma unroll
        for (int j = 0; j < 2; ++j) {
            int mb = wm + i * 16 + lhi * 8;          // 8 consecutive channels
            int ng = n0 + wn + j * 16 + l16;
            union { float f[8]; u32x4 v[2]; } yv;
            union { unsigned short s[8]; u32x4 v; } yb;
            #pragma unroll
            for (int r = 0; r < 8; ++r) {
                float v1 = acc[0][i][j][r] + b1[mb + r];
                v1 = v1 >= 0.f ? v1 : alpha * v1;
                float v4 = acc[1][i][j][r] + b4[mb + r];
                v4 = v4 >= 0.f ? v4 : alpha * v4;
                float y = v1 + v4;
                yv.f[r] = y;
                yb.s[r] = f2bf_bits(y);
            }
            size_t ob = ((size_t)b * HWC + ng) * 128 + mb;   // NHWC
            *(u32x4*)(y1 + ob)     = yv.v[0];
            *(u32x4*)(y1 + ob + 4) = yv.v[1];
            *(u32x4*)((unsigned short*)y1b + ob) = yb.v;
        }
    }
}

// ---------------------------------------------------------------------------
// 2) 1x1-conv GEMM, NHWC: Out[b][n][m] = epi(sum_k W[m][k] X[b][n][k] + bias).
//    Tiles staged by TDM (hardware-padded rows) when available.
// ---------------------------------------------------------------------------
enum {
    EP_TANH_BF16 = 0,
    EP_TANH_F32,
    EP_SIG_F32,
    EP_PSA_BF16,     // v*y1 + y1          (aux = y1 f32 NHWC, C=128)
    EP_GELU_BF16,
    EP_FINAL_F32     // v + ygate -> NCHW  (aux = ygate f32 NHWC, C=128)
};

template<int EPI, int K>
__global__ __launch_bounds__(256)
void gemm_bias_act_kernel(const __hip_bfloat16* __restrict__ Wb, // [M_CH][K]
                          const float* __restrict__ bias,        // [M_CH]
                          const __hip_bfloat16* __restrict__ X,  // [B][HW][K]
                          void* __restrict__ Out,
                          const float* __restrict__ aux,
                          int M_CH)
{
    constexpr int KP = K + 8;
    extern __shared__ char smem[];
    __hip_bfloat16* Al = (__hip_bfloat16*)smem;   // [128][KP] weights
    __hip_bfloat16* Bl = Al + 128 * KP;           // [128][KP] activations

    const int tid = threadIdx.x;
    const int n0  = blockIdx.x * 128;
    const int b   = blockIdx.y;
    const int m0  = blockIdx.z * 128;

#ifdef HAVE_TDM
    if (tid < 32) {   // wave 0 issues both DMA descriptors (wave-uniform)
        tdm_load_tile_pad16<K, 128>(Wb + (size_t)m0 * K,
                                    (unsigned)(unsigned long long)(size_t)Al);
        tdm_load_tile_pad16<K, 128>(X + ((size_t)b * HWC + n0) * K,
                                    (unsigned)(unsigned long long)(size_t)Bl);
        __builtin_amdgcn_s_wait_tensorcnt(0);
    }
#else
    for (int idx = tid; idx < 128 * K; idx += 256) {
        int r = idx / K, k = idx % K;
        Al[r * KP + k] = Wb[(size_t)(m0 + r) * K + k];
    }
    for (int idx = tid; idx < 128 * K; idx += 256) {
        int n = idx / K, k = idx % K;
        Bl[n * KP + k] = X[((size_t)b * HWC + n0 + n) * K + k];
    }
#endif
    __syncthreads();

    const int lane   = tid & 31;
    const int waveId = tid >> 5;
    const int wm     = (waveId & 3) * 32;
    const int wn     = (waveId >> 2) * 64;
    const int l16    = lane & 15;
    const int lhi    = lane >> 4;

    v8f acc[2][4] = {};
    #pragma unroll
    for (int k0 = 0; k0 < K; k0 += 32) {
        Frag a[2], bf[4];
        #pragma unroll
        for (int i = 0; i < 2; ++i) {
            const __hip_bfloat16* p = &Al[(wm + i * 16 + l16) * KP + k0 + lhi * 8];
            a[i].u[0] = *(const u32x4*)(p);
            a[i].u[1] = *(const u32x4*)(p + 16);
        }
        #pragma unroll
        for (int j = 0; j < 4; ++j) {
            const __hip_bfloat16* p = &Bl[(wn + j * 16 + l16) * KP + k0 + lhi * 8];
            bf[j].u[0] = *(const u32x4*)(p);
            bf[j].u[1] = *(const u32x4*)(p + 16);
        }
        #pragma unroll
        for (int i = 0; i < 2; ++i)
            #pragma unroll
            for (int j = 0; j < 4; ++j)
                acc[i][j] = __builtin_amdgcn_wmma_f32_16x16x32_bf16(
                    false, a[i].v, false, bf[j].v, (short)0, acc[i][j],
                    false, false);
    }

    #pragma unroll
    for (int i = 0; i < 2; ++i) {
        #pragma unroll
        for (int j = 0; j < 4; ++j) {
            int mb = m0 + wm + i * 16 + lhi * 8;     // 8 consecutive channels
            int ng = n0 + wn + j * 16 + l16;
            float v[8];
            #pragma unroll
            for (int r = 0; r < 8; ++r) v[r] = acc[i][j][r] + bias[mb + r];

            if (EPI == EP_FINAL_F32) {               // NCHW output, strided
                #pragma unroll
                for (int r = 0; r < 8; ++r) {
                    float o = v[r] + aux[((size_t)b * HWC + ng) * 128 + mb + r];
                    ((float*)Out)[((size_t)b * 128 + mb + r) * HWC + ng] = o;
                }
            } else if (EPI == EP_TANH_F32 || EPI == EP_SIG_F32) {
                union { float f[8]; u32x4 q[2]; } ov;
                #pragma unroll
                for (int r = 0; r < 8; ++r)
                    ov.f[r] = (EPI == EP_TANH_F32)
                                  ? tanhf(v[r])
                                  : 1.f / (1.f + __expf(-v[r]));
                size_t ob = ((size_t)b * HWC + ng) * M_CH + mb;
                *(u32x4*)((float*)Out + ob)     = ov.q[0];
                *(u32x4*)((float*)Out + ob + 4) = ov.q[1];
            } else {                                 // bf16 NHWC outputs
                union { unsigned short s[8]; u32x4 q; } ov;
                #pragma unroll
                for (int r = 0; r < 8; ++r) {
                    float o;
                    if (EPI == EP_TANH_BF16) {
                        o = tanhf(v[r]);
                    } else if (EPI == EP_PSA_BF16) {
                        float a0 = aux[((size_t)b * HWC + ng) * 128 + mb + r];
                        o = v[r] * a0 + a0;
                    } else {                         // EP_GELU_BF16
                        o = 0.5f * v[r] * (1.f + erff(v[r] * 0.70710678118f));
                    }
                    ov.s[r] = f2bf_bits(o);
                }
                size_t ob = ((size_t)b * HWC + ng) * M_CH + mb;
                *(u32x4*)((unsigned short*)Out + ob) = ov.q;
            }
        }
    }
}

// ---------------------------------------------------------------------------
// 3) SRU scan over H, NHWC. Threads channel-major: wave = 32 consecutive
//    channels -> fwd/rev direction (split at c=64) is wave-uniform.
//    Fused gate: ygate = y1 * sigmoid(h_state).
// ---------------------------------------------------------------------------
__global__ __launch_bounds__(256)
void scan_gate_kernel(const float* __restrict__ Z,     // [B][HW][128]
                      const float* __restrict__ Fg,
                      const float* __restrict__ y1,
                      float* __restrict__ ygate)
{
    const int c = threadIdx.x & 127;
    const int w = blockIdx.x * 2 + (threadIdx.x >> 7);
    const int b = blockIdx.y;
    const size_t base = ((size_t)b * HWC + w) * 128 + c;   // (b, h=0, w, c)
    const size_t hstep = (size_t)IMG_W * 128;
    float hs = 0.f;
    if (c >= 64) {                       // forward scan (flip-unflip algebra)
        for (int h = 0; h < IMG_H; ++h) {
            size_t idx = base + (size_t)h * hstep;
            float z = Z[idx], f = Fg[idx];
            hs = f * hs + (1.f - f) * z;
            ygate[idx] = y1[idx] * (1.f / (1.f + __expf(-hs)));
        }
    } else {                             // reverse scan
        for (int h = IMG_H - 1; h >= 0; --h) {
            size_t idx = base + (size_t)h * hstep;
            float z = Z[idx], f = Fg[idx];
            hs = f * hs + (1.f - f) * z;
            ygate[idx] = y1[idx] * (1.f / (1.f + __expf(-hs)));
        }
    }
}

// ---------------------------------------------------------------------------
// Host orchestration
// ---------------------------------------------------------------------------
extern "C" void kernel_launch(void* const* d_in, const int* in_sizes, int n_in,
                              void* d_out, int out_size, void* d_ws, size_t ws_size,
                              hipStream_t stream)
{
    const float* x    = (const float*)d_in[0];
    const float* d1w  = (const float*)d_in[1];
    const float* d1b  = (const float*)d_in[2];
    const float* d4w  = (const float*)d_in[3];
    const float* d4b  = (const float*)d_in[4];
    const float* pa   = (const float*)d_in[5];
    const float* f1w1 = (const float*)d_in[6];
    const float* f1b1 = (const float*)d_in[7];
    const float* f1w2 = (const float*)d_in[8];
    const float* f1b2 = (const float*)d_in[9];
    const float* f2w1 = (const float*)d_in[10];
    const float* f2b1 = (const float*)d_in[11];
    const float* f2w2 = (const float*)d_in[12];
    const float* f2b2 = (const float*)d_in[13];
    const float* pw1  = (const float*)d_in[14];
    const float* pb1  = (const float*)d_in[15];
    const float* pw2  = (const float*)d_in[16];
    const float* pb2  = (const float*)d_in[17];
    const float* pw3  = (const float*)d_in[18];
    const float* pb3  = (const float*)d_in[19];

    char* ws = (char*)d_ws;
    const size_t szF  = (size_t)NB * 128 * HWC * 4;   // 32 MB
    const size_t szBf = (size_t)NB * 128 * HWC * 2;   // 16 MB

    float*          y1    = (float*)(ws);                       // NHWC f32
    __hip_bfloat16* y1b   = (__hip_bfloat16*)(ws + szF);        // NHWC bf16
    float*          ygate = (float*)(ws + szF + szBf);
    float*          Z     = (float*)(ws + 2 * szF + szBf);
    float*          Fg    = (float*)(ws + 3 * szF + szBf);
    __hip_bfloat16* t     = (__hip_bfloat16*)(ws + 4 * szF + szBf);
    __hip_bfloat16* g     = (__hip_bfloat16*)Z;   // [B][HW][256], reuses Z (32 MB)
    __hip_bfloat16* wbase = (__hip_bfloat16*)(ws + 4 * szF + 2 * szBf);
    __hip_bfloat16* wb_f1w1 = wbase;
    __hip_bfloat16* wb_f1w2 = wbase + 16384;
    __hip_bfloat16* wb_f2w1 = wbase + 2 * 16384;
    __hip_bfloat16* wb_f2w2 = wbase + 3 * 16384;
    __hip_bfloat16* wb_pw3  = wbase + 4 * 16384;
    __hip_bfloat16* wb_pw1  = wbase + 5 * 16384;            // 256x128
    __hip_bfloat16* wb_pw2  = wbase + 5 * 16384 + 32768;    // 128x256

    // 0) weight conversion (tiny)
    cvt_f32_bf16_kernel<<<64, 256, 0, stream>>>(f1w1, wb_f1w1, 16384);
    cvt_f32_bf16_kernel<<<64, 256, 0, stream>>>(f1w2, wb_f1w2, 16384);
    cvt_f32_bf16_kernel<<<64, 256, 0, stream>>>(f2w1, wb_f2w1, 16384);
    cvt_f32_bf16_kernel<<<64, 256, 0, stream>>>(f2w2, wb_f2w2, 16384);
    cvt_f32_bf16_kernel<<<64, 256, 0, stream>>>(pw3, wb_pw3, 16384);
    cvt_f32_bf16_kernel<<<128, 256, 0, stream>>>(pw1, wb_pw1, 32768);
    cvt_f32_bf16_kernel<<<128, 256, 0, stream>>>(pw2, wb_pw2, 32768);

    // 1) DPB conv
    {
        dim3 grid(HWC / 64, NB);
        size_t sh = (size_t)(2 * 128 * 104 + 2 * 64 * 104) * 2;
        conv_dpb_kernel<<<grid, 256, sh, stream>>>(
            x, d1w, d1b, d4w, d4b, pa, y1, y1b);
    }

    const size_t sh128 = (size_t)2 * 128 * (128 + 8) * 2;   // ~68 KB
    const size_t sh256 = (size_t)2 * 128 * (256 + 8) * 2;   // ~132 KB
    dim3 g128 (HWC / 128, NB, 1);
    dim3 g256m(HWC / 128, NB, 2);

    // 2-5) gate inputs
    gemm_bias_act_kernel<EP_TANH_BF16, 128><<<g128, 256, sh128, stream>>>(
        wb_f1w1, f1b1, y1b, t, nullptr, 128);
    gemm_bias_act_kernel<EP_TANH_F32, 128><<<g128, 256, sh128, stream>>>(
        wb_f1w2, f1b2, t, Z, nullptr, 128);
    gemm_bias_act_kernel<EP_TANH_BF16, 128><<<g128, 256, sh128, stream>>>(
        wb_f2w1, f2b1, y1b, t, nullptr, 128);
    gemm_bias_act_kernel<EP_SIG_F32, 128><<<g128, 256, sh128, stream>>>(
        wb_f2w2, f2b2, t, Fg, nullptr, 128);

    // 6) scan + gate
    scan_gate_kernel<<<dim3(IMG_W / 2, NB), 256, 0, stream>>>(Z, Fg, y1, ygate);

    // 7-9) PSA + final combine
    gemm_bias_act_kernel<EP_PSA_BF16, 128><<<g128, 256, sh128, stream>>>(
        wb_pw3, pb3, y1b, t, y1, 128);
    gemm_bias_act_kernel<EP_GELU_BF16, 128><<<g256m, 256, sh128, stream>>>(
        wb_pw1, pb1, t, g, nullptr, 256);
    gemm_bias_act_kernel<EP_FINAL_F32, 256><<<g128, 256, sh256, stream>>>(
        wb_pw2, pb2, g, (float*)d_out, ygate, 128);
}
